// MambaDecoder_62792421867550
// MI455X (gfx1250) — compile-verified, matrix operand order checked by reading, not run on hardware
//
#include <hip/hip_runtime.h>
#include <hip/hip_bf16.h>
#include <cstddef>

// ---------------------------------------------------------------------------
// MambaDecoder forward for MI455X (gfx1250, wave32, WMMA).
// All dense projections run through v_wmma_f32_16x16x32_f16 (f16 in, f32 acc).
// ---------------------------------------------------------------------------

#define BATCH   32
#define T_LEN   512
#define NTOK    (BATCH * T_LEN)       // 16384
#define LATENT  256
#define DIM_I   512                   // LATENT * EXPAND
#define DSTATE  16
#define DTRANK  16
#define DCONV   4
#define NGROUPS 16
#define OUTDIM  120

typedef __attribute__((ext_vector_type(16))) _Float16 v16h;
typedef __attribute__((ext_vector_type(8)))  _Float16 h8v;
typedef __attribute__((ext_vector_type(8)))  float    v8f;
typedef __attribute__((ext_vector_type(4)))  float    f4v;

union AFrag { v16h v; h8v h[2]; };

// ------------------------------ WMMA GEMM ----------------------------------
// out[M,N] = act( A[M,K](lda) @ W[N,K]^T(ldw) + bias )
// act: 0=none, 1=silu, 2=softplus
// Block tile 64(M) x 64(N) x 64(K); 8 waves as 4Mx2N; wave tile 16x32;
// 4 WMMAs per wave per LDS stage; LDS double-buffered so next-stage global
// loads overlap current-stage WMMAs with a single barrier per stage.
// FULL=true: M,N,K multiples of the tile -> all bounds checks removed.
#define GBM 64
#define GBN 64
#define GBK 64
#define SPITCH 72   // LDS row pitch in halves; 144B keeps 16B alignment

template <bool FULL>
__global__ __launch_bounds__(256)
void gemm_wmma(const float* __restrict__ A, int lda,
               const float* __restrict__ W, int ldw,
               const float* __restrict__ bias,
               float* __restrict__ out, int ldo,
               int M, int N, int K, int act)
{
    __shared__ __attribute__((aligned(16))) _Float16 sA[2][GBM * SPITCH];
    __shared__ __attribute__((aligned(16))) _Float16 sB[2][GBN * SPITCH];

    const int tid  = threadIdx.x;
    const int wave = tid >> 5;
    const int lane = tid & 31;
    const int wm   = wave & 3;      // M subtile 0..3
    const int wn   = wave >> 2;     // N slab 0..1 (32 cols each)
    const int m0   = blockIdx.y * GBM;
    const int n0   = blockIdx.x * GBN;
    const int col  = lane & 15;
    const int hi   = lane >> 4;     // half-wave selects K groups

    v8f acc[2] = {{}, {}};

    // staging coords: each thread owns one row x 16 k of each tile
    const int sr = tid >> 2;          // 0..63
    const int kc = (tid & 3) << 4;    // 0,16,32,48
    const float* srcA = A + (size_t)(m0 + sr) * lda + kc;
    const float* srcW = W + (size_t)(n0 + sr) * ldw + kc;
    const bool okA = FULL || (m0 + sr < M);
    const bool okW = FULL || (n0 + sr < N);

    auto load_tile = [&](const float* src, bool rowok, int k0, f4v r[4]) {
        #pragma unroll
        for (int c = 0; c < 2; ++c) {
            f4v f0 = {}, f1 = {};
            if (FULL || (rowok && (k0 + kc + c * 8) < K)) {  // K%8==0 -> uniform
                f0 = *(const f4v*)(src + k0 + c * 8);
                f1 = *(const f4v*)(src + k0 + c * 8 + 4);
            }
            r[2 * c] = f0; r[2 * c + 1] = f1;
        }
    };
    auto stash_tile = [&](const f4v r[4], _Float16* base) {
        #pragma unroll
        for (int c = 0; c < 2; ++c) {
            h8v h;
            h[0] = (_Float16)r[2 * c][0]; h[1] = (_Float16)r[2 * c][1];
            h[2] = (_Float16)r[2 * c][2]; h[3] = (_Float16)r[2 * c][3];
            h[4] = (_Float16)r[2 * c + 1][0]; h[5] = (_Float16)r[2 * c + 1][1];
            h[6] = (_Float16)r[2 * c + 1][2]; h[7] = (_Float16)r[2 * c + 1][3];
            ((h8v*)base)[c] = h;
        }
    };

    const int nstages = (K + GBK - 1) / GBK;

    // prologue: stage 0 into buffer 0
    {
        f4v ra[4], rw[4];
        load_tile(srcA, okA, 0, ra);
        load_tile(srcW, okW, 0, rw);
        stash_tile(ra, &sA[0][sr * SPITCH + kc]);
        stash_tile(rw, &sB[0][sr * SPITCH + kc]);
    }
    __syncthreads();

    for (int st = 0; st < nstages; ++st) {
        const int cur = st & 1;
        const int nxt = cur ^ 1;
        const bool more = (st + 1) < nstages;

        // issue next-stage global loads before consuming current LDS stage
        f4v ra[4], rw[4];
        if (more) {
            const int k0n = (st + 1) * GBK;
            load_tile(srcA, okA, k0n, ra);
            load_tile(srcW, okW, k0n, rw);
            if (st + 2 < nstages) {
                __builtin_prefetch(srcA + k0n + GBK, 0, 3);
                __builtin_prefetch(srcW + k0n + GBK, 0, 3);
            }
        }

        // ---- fragments per CDNA5 16-bit lane layout:
        // lane<16 holds K = {0..7, 16..23}; lane>=16 holds K = {8..15, 24..31}
        AFrag a[2], b[2][2];
        {
            const int arow = (wm * 16 + col) * SPITCH;
            #pragma unroll
            for (int s = 0; s < 2; ++s) {
                a[s].h[0] = *(const h8v*)&sA[cur][arow + s * 32 + hi * 8];
                a[s].h[1] = *(const h8v*)&sA[cur][arow + s * 32 + 16 + hi * 8];
            }
            #pragma unroll
            for (int ns = 0; ns < 2; ++ns) {
                const int brow = (wn * 32 + ns * 16 + col) * SPITCH;
                #pragma unroll
                for (int s = 0; s < 2; ++s) {
                    b[ns][s].h[0] = *(const h8v*)&sB[cur][brow + s * 32 + hi * 8];
                    b[ns][s].h[1] = *(const h8v*)&sB[cur][brow + s * 32 + 16 + hi * 8];
                }
            }
        }
        // alternate accumulators so consecutive WMMAs are independent
        #pragma unroll
        for (int s = 0; s < 2; ++s)
            #pragma unroll
            for (int ns = 0; ns < 2; ++ns)
                acc[ns] = __builtin_amdgcn_wmma_f32_16x16x32_f16(
                              false, a[s].v, false, b[ns][s].v,
                              (short)0, acc[ns], false, false);

        // convert + store next stage into the other buffer
        if (more) {
            stash_tile(ra, &sA[nxt][sr * SPITCH + kc]);
            stash_tile(rw, &sB[nxt][sr * SPITCH + kc]);
        }
        __syncthreads();
    }

    // ---- epilogue: C layout = VGPR r -> M = r + 8*(lane>=16), N = lane&15
    #pragma unroll
    for (int ns = 0; ns < 2; ++ns) {
        const int gn = n0 + wn * 32 + ns * 16 + col;
        #pragma unroll
        for (int r = 0; r < 8; ++r) {
            const int gm = m0 + wm * 16 + hi * 8 + r;
            if (FULL || (gm < M && gn < N)) {
                float v = acc[ns][r];
                if (bias) v += bias[gn];
                if (act == 1)      v = v / (1.f + __expf(-v));                   // silu
                else if (act == 2) v = (v > 20.f) ? v : __logf(1.f + __expf(v)); // softplus
                out[(size_t)gm * ldo + gn] = v;
            }
        }
    }
}

// --------------------------- LayerNorm (C=256) -----------------------------
__global__ __launch_bounds__(256)
void layernorm_kernel(const float* __restrict__ in,
                      const float* __restrict__ g,
                      const float* __restrict__ b,
                      float* __restrict__ out, int act_silu)
{
    __shared__ float red[256];
    const int tid = threadIdx.x;
    const size_t tok = blockIdx.x;
    const float x = in[tok * LATENT + tid];

    red[tid] = x; __syncthreads();
    #pragma unroll
    for (int s = 128; s > 0; s >>= 1) {
        if (tid < s) red[tid] += red[tid + s];
        __syncthreads();
    }
    const float m = red[0] * (1.f / LATENT);
    __syncthreads();

    const float d = x - m;
    red[tid] = d * d; __syncthreads();
    #pragma unroll
    for (int s = 128; s > 0; s >>= 1) {
        if (tid < s) red[tid] += red[tid + s];
        __syncthreads();
    }
    const float var = red[0] * (1.f / LATENT);

    float y = d * rsqrtf(var + 1e-5f) * g[tid] + b[tid];
    if (act_silu) y = y / (1.f + __expf(-y));
    out[tok * LATENT + tid] = y;
}

// -------------------------- z + pos_embed broadcast ------------------------
__global__ __launch_bounds__(256)
void add_pos_kernel(const float* __restrict__ zlat,
                    const float* __restrict__ pos,
                    float* __restrict__ x)
{
    const size_t idx = (size_t)blockIdx.x * blockDim.x + threadIdx.x;
    if (idx >= (size_t)NTOK * LATENT) return;
    const int c   = idx & (LATENT - 1);
    const int tok = idx / LATENT;
    const int b   = tok / T_LEN;
    const int t   = tok - b * T_LEN;
    x[idx] = zlat[b * LATENT + c] + pos[(size_t)t * LATENT + c];
}

// -------- generic time-flip / strided copy (for bwd dir and concat) --------
__global__ __launch_bounds__(256)
void copy_time_kernel(const float* __restrict__ src, float* __restrict__ dst,
                      int C, int ldd, int dstoff, int flip)
{
    const size_t idx = (size_t)blockIdx.x * blockDim.x + threadIdx.x;
    if (idx >= (size_t)NTOK * C) return;
    const int c   = idx % C;
    const int tok = idx / C;
    const int b   = tok / T_LEN;
    const int t   = tok - b * T_LEN;
    const int st  = flip ? (T_LEN - 1 - t) : t;
    dst[(size_t)tok * ldd + dstoff + c] = src[(size_t)(b * T_LEN + st) * C + c];
}

// ----------------- depthwise causal conv (k=4) + bias + silu ---------------
// xz layout [tok][1024]; channels 0..511 are the conv input. out xc [tok][512]
__global__ __launch_bounds__(256)
void conv_dw_kernel(const float* __restrict__ xz,
                    const float* __restrict__ w,   // [512][1][4]
                    const float* __restrict__ bias,
                    float* __restrict__ xc)
{
    const size_t idx = (size_t)blockIdx.x * blockDim.x + threadIdx.x;
    if (idx >= (size_t)NTOK * DIM_I) return;
    const int d   = idx & (DIM_I - 1);
    const int tok = idx / DIM_I;
    const int b   = tok / T_LEN;
    const int t   = tok - b * T_LEN;

    float acc = bias[d];
    #pragma unroll
    for (int j = 0; j < DCONV; ++j) {
        const int tt = t + j - (DCONV - 1);
        if (tt >= 0)
            acc += w[d * DCONV + j] * xz[(size_t)(b * T_LEN + tt) * (2 * DIM_I) + d];
    }
    xc[idx] = acc / (1.f + __expf(-acc));   // silu
}

// ----------------------------- selective scan ------------------------------
// Sequential in T; parallel over (batch, channel). h[16] lives in registers.
// Fused: y = (scan + u*D) * silu(zg).
__global__ __launch_bounds__(256)
void scan_kernel(const float* __restrict__ xc,    // u   [tok][512]
                 const float* __restrict__ xz,    // zg = [tok][1024], cols 512..
                 const float* __restrict__ dtb,   // dt  [tok][512]
                 const float* __restrict__ xdbl,  // [tok][48]: dt raw|B|C
                 const float* __restrict__ A_log, // [512][16]
                 const float* __restrict__ Dp,    // [512]
                 float* __restrict__ yout)        // [tok][512]
{
    const int tid = threadIdx.x;
    const int b   = blockIdx.x >> 1;
    const int d   = ((blockIdx.x & 1) << 8) + tid;

    float Arow[DSTATE];
    #pragma unroll
    for (int n = 0; n < DSTATE; ++n) Arow[n] = -__expf(A_log[d * DSTATE + n]);
    const float Dd = Dp[d];

    float h[DSTATE];
    #pragma unroll
    for (int n = 0; n < DSTATE; ++n) h[n] = 0.f;

    __shared__ float sBC[32];   // B_t[16] | C_t[16]

    for (int t = 0; t < T_LEN; ++t) {
        const size_t tok = (size_t)b * T_LEN + t;
        if (tid < 32) sBC[tid] = xdbl[tok * 48 + DTRANK + tid];
        __syncthreads();

        const float dtv = dtb[tok * DIM_I + d];
        const float u   = xc[tok * DIM_I + d];
        const float du  = dtv * u;
        float y = 0.f;
        #pragma unroll
        for (int n = 0; n < DSTATE; ++n) {
            h[n] = h[n] * __expf(dtv * Arow[n]) + du * sBC[n];
            y += h[n] * sBC[DSTATE + n];
        }
        const float zg = xz[tok * (2 * DIM_I) + DIM_I + d];
        yout[tok * DIM_I + d] = (y + u * Dd) * (zg / (1.f + __expf(-zg)));
        __syncthreads();
    }
}

// ------------------------- GroupNorm over (D/G, T) -------------------------
__global__ __launch_bounds__(256)
void groupnorm_kernel(const float* __restrict__ x,
                      const float* __restrict__ g,
                      const float* __restrict__ b,
                      float* __restrict__ out)
{
    const int tid = threadIdx.x;
    const int bb  = blockIdx.x >> 4;
    const int grp = blockIdx.x & 15;
    const int CG  = LATENT / NGROUPS;         // 16
    const int cnt = CG * T_LEN;               // 8192

    __shared__ float s1[256], s2[256];
    float sum = 0.f, sq = 0.f;
    for (int i = tid; i < cnt; i += 256) {
        const int t  = i >> 4;
        const int cc = i & 15;
        const float v = x[(size_t)(bb * T_LEN + t) * LATENT + grp * CG + cc];
        sum += v; sq += v * v;
    }
    s1[tid] = sum; s2[tid] = sq; __syncthreads();
    #pragma unroll
    for (int s = 128; s > 0; s >>= 1) {
        if (tid < s) { s1[tid] += s1[tid + s]; s2[tid] += s2[tid + s]; }
        __syncthreads();
    }
    const float m   = s1[0] / cnt;
    const float var = s2[0] / cnt - m * m;
    const float rs  = rsqrtf(var + 1e-5f);

    for (int i = tid; i < cnt; i += 256) {
        const int t  = i >> 4;
        const int cc = i & 15;
        const int c  = grp * CG + cc;
        const size_t off = (size_t)(bb * T_LEN + t) * LATENT + c;
        out[off] = (x[off] - m) * rs * g[c] + b[c];
    }
}

// ---------------- im2col for 1D conv k=3 pad=1, K-layout ci*3+j -----------
__global__ __launch_bounds__(256)
void im2col_kernel(const float* __restrict__ xg, float* __restrict__ col)
{
    const size_t idx = (size_t)blockIdx.x * blockDim.x + threadIdx.x;
    if (idx >= (size_t)NTOK * 768) return;
    const int k   = idx % 768;
    const int tok = idx / 768;
    const int b   = tok / T_LEN;
    const int t   = tok - b * T_LEN;
    const int ci  = k / 3;
    const int j   = k - ci * 3;
    const int tt  = t + j - 1;
    col[idx] = (tt >= 0 && tt < T_LEN)
                 ? xg[(size_t)(b * T_LEN + tt) * LATENT + ci] : 0.f;
}

// -------------------- x = (x + delta) * mask (in place) --------------------
__global__ __launch_bounds__(256)
void residual_mask_kernel(float* __restrict__ x, const float* __restrict__ delta,
                          const int* __restrict__ lengths)
{
    const size_t idx = (size_t)blockIdx.x * blockDim.x + threadIdx.x;
    if (idx >= (size_t)NTOK * LATENT) return;
    const int tok = idx / LATENT;
    const int b   = tok / T_LEN;
    const int t   = tok - b * T_LEN;
    const float mask = (t < lengths[b]) ? 1.f : 0.f;
    x[idx] = (x[idx] + delta[idx]) * mask;
}

__global__ __launch_bounds__(256)
void mask_out_kernel(float* __restrict__ o, const int* __restrict__ lengths)
{
    const size_t idx = (size_t)blockIdx.x * blockDim.x + threadIdx.x;
    if (idx >= (size_t)NTOK * OUTDIM) return;
    const int tok = idx / OUTDIM;
    const int b   = tok / T_LEN;
    const int t   = tok - b * T_LEN;
    if (t >= lengths[b]) o[idx] = 0.f;
}

// ===========================================================================
extern "C" void kernel_launch(void* const* d_in, const int* in_sizes, int n_in,
                              void* d_out, int out_size, void* d_ws, size_t ws_size,
                              hipStream_t stream)
{
    (void)in_sizes; (void)n_in; (void)out_size; (void)ws_size;

    // ---- input leaf order: recursive insertion-order flatten of setup_inputs
    // 0:z  1:latent_w 2:latent_b 3:latent_ln_g 4:latent_ln_b 5:pos_embed
    // per layer L (base = 6 + 26*L):
    //   +0 ln_g +1 ln_b
    //   fwd: +2 in_proj +3 conv_w +4 conv_b +5 x_proj +6 dt_w +7 dt_b
    //        +8 A_log +9 D +10 out_proj
    //   bwd: +11..+19 (same order)
    //   +20 comb_w +21 comb_b +22 gn_g +23 gn_b +24 lconv_w +25 lconv_b
    // 110 final_ln_g 111 final_ln_b 112 out_w 113 out_b 114 lengths
    #define PIN(i) ((const float*)d_in[(i)])
    const int after_layers = 6 + 4 * 26;                  // 110
    const int* lengths = (const int*)d_in[after_layers + 4];

    // ---- workspace arena (floats)
    float* ws = (float*)d_ws;
    float* X    = ws;                               // 16384*256
    float* XN   = X    + (size_t)NTOK * 256;        // 16384*256
    float* XNF  = XN   + (size_t)NTOK * 256;        // 16384*256
    float* XZ   = XNF  + (size_t)NTOK * 256;        // 16384*1024 (reused as im2col)
    float* XC   = XZ   + (size_t)NTOK * 1024;       // 16384*512
    float* DT   = XC   + (size_t)NTOK * 512;        // 16384*512
    float* XDBL = DT   + (size_t)NTOK * 512;        // 16384*48
    float* Y    = XDBL + (size_t)NTOK * 48;         // 16384*512
    float* DIR  = Y    + (size_t)NTOK * 512;        // 16384*256
    float* FB   = DIR  + (size_t)NTOK * 256;        // 16384*512
    float* TMP  = FB   + (size_t)NTOK * 512;        // 16384*256
    float* ZL   = TMP  + (size_t)NTOK * 256;        // 32*256
    float* ZL2  = ZL   + (size_t)BATCH * 256;       // 32*256

    auto gemm = [&](const float* A, int lda, const float* W, int ldw,
                    const float* bias, float* out, int ldo,
                    int M, int N, int K, int act) {
        dim3 grid((N + GBN - 1) / GBN, (M + GBM - 1) / GBM);
        const bool full = (M % GBM == 0) && (N % GBN == 0) && (K % GBK == 0);
        if (full)
            gemm_wmma<true><<<grid, 256, 0, stream>>>(A, lda, W, ldw, bias, out, ldo, M, N, K, act);
        else
            gemm_wmma<false><<<grid, 256, 0, stream>>>(A, lda, W, ldw, bias, out, ldo, M, N, K, act);
    };
    auto eltblocks = [](size_t n) { return (unsigned)((n + 255) / 256); };

    // ---------------- latent stem ----------------
    gemm(PIN(0), LATENT, PIN(1), LATENT, PIN(2), ZL, LATENT, BATCH, LATENT, LATENT, 0);
    layernorm_kernel<<<BATCH, 256, 0, stream>>>(ZL, PIN(3), PIN(4), ZL2, /*silu*/1);
    add_pos_kernel<<<eltblocks((size_t)NTOK * 256), 256, 0, stream>>>(ZL2, PIN(5), X);

    // ---------------- mamba direction pipeline ----------------
    auto run_dir = [&](const float* xnbuf, int pb, int half, int flip) {
        const float* in_proj = PIN(pb + 0);
        const float* conv_w  = PIN(pb + 1);
        const float* conv_b  = PIN(pb + 2);
        const float* x_proj  = PIN(pb + 3);
        const float* dt_w    = PIN(pb + 4);
        const float* dt_b    = PIN(pb + 5);
        const float* A_log   = PIN(pb + 6);
        const float* Dp      = PIN(pb + 7);
        const float* out_prj = PIN(pb + 8);

        gemm(xnbuf, 256, in_proj, 256, nullptr, XZ, 1024, NTOK, 1024, 256, 0);
        conv_dw_kernel<<<eltblocks((size_t)NTOK * 512), 256, 0, stream>>>(XZ, conv_w, conv_b, XC);
        gemm(XC, 512, x_proj, 512, nullptr, XDBL, 48, NTOK, 48, 512, 0);
        gemm(XDBL, 48, dt_w, 16, dt_b, DT, 512, NTOK, 512, 16, /*softplus*/2);
        scan_kernel<<<BATCH * 2, 256, 0, stream>>>(XC, XZ, DT, XDBL, A_log, Dp, Y);
        gemm(Y, 512, out_prj, 512, nullptr, DIR, 256, NTOK, 256, 512, 0);
        copy_time_kernel<<<eltblocks((size_t)NTOK * 256), 256, 0, stream>>>(
            DIR, FB, 256, 512, half * 256, flip);
    };

    // ---------------- layers ----------------
    for (int L = 0; L < 4; ++L) {
        const int lb = 6 + L * 26;

        layernorm_kernel<<<NTOK, 256, 0, stream>>>(X, PIN(lb + 0), PIN(lb + 1), XN, 0);
        run_dir(XN, lb + 2, 0, 0);
        copy_time_kernel<<<eltblocks((size_t)NTOK * 256), 256, 0, stream>>>(
            XN, XNF, 256, 256, 0, /*flip*/1);
        run_dir(XNF, lb + 11, 1, 1);

        // comb projection + residual + mask
        gemm(FB, 512, PIN(lb + 20), 512, PIN(lb + 21), TMP, 256, NTOK, 256, 512, 0);
        residual_mask_kernel<<<eltblocks((size_t)NTOK * 256), 256, 0, stream>>>(X, TMP, lengths);

        // local module: groupnorm -> im2col -> WMMA conv GEMM (+silu) -> residual
        groupnorm_kernel<<<BATCH * NGROUPS, 256, 0, stream>>>(X, PIN(lb + 22), PIN(lb + 23), TMP);
        im2col_kernel<<<eltblocks((size_t)NTOK * 768), 256, 0, stream>>>(TMP, XZ);
        gemm(XZ, 768, PIN(lb + 24), 768, PIN(lb + 25), DIR, 256, NTOK, 256, 768, /*silu*/1);
        residual_mask_kernel<<<eltblocks((size_t)NTOK * 256), 256, 0, stream>>>(X, DIR, lengths);
    }

    // ---------------- head ----------------
    layernorm_kernel<<<NTOK, 256, 0, stream>>>(X, PIN(after_layers + 0), PIN(after_layers + 1), XN, 0);
    gemm(XN, 256, PIN(after_layers + 2), 256, PIN(after_layers + 3),
         (float*)d_out, OUTDIM, NTOK, OUTDIM, LATENT, 0);
    mask_out_kernel<<<eltblocks((size_t)NTOK * OUTDIM), 256, 0, stream>>>((float*)d_out, lengths);
}
